// MoEBlock_43069932044301
// MI455X (gfx1250) — compile-verified, matrix operand order checked by reading, not run on hardware
//
#include <hip/hip_runtime.h>
#include <hip/hip_bf16.h>

#define T_TOK   16384
#define DMODEL  1024
#define HDIM    4096
#define NEXP    8
#define CAP     1024

#define XS_STRIDE 1032   // 1024 + 8 pad (bf16 elems)
#define HS_STRIDE 4104   // 4096 + 8 pad

typedef __attribute__((ext_vector_type(16))) __bf16    v16bf;
typedef __attribute__((ext_vector_type(8)))  float     v8f;
typedef __attribute__((ext_vector_type(4)))  unsigned  v4u;
typedef __attribute__((ext_vector_type(8)))  unsigned  v8u;

// ---------------------------------------------------------------- router ----
__global__ __launch_bounds__(256)
void router_kernel(const float* __restrict__ x, const float* __restrict__ wsw,
                   const float* __restrict__ bsw, int* __restrict__ routes) {
    const int wid  = (blockIdx.x * blockDim.x + threadIdx.x) >> 5; // 1 wave / token
    const int lane = threadIdx.x & 31;
    if (wid >= T_TOK) return;
    const float* xr = x + (size_t)wid * DMODEL;
    float acc[NEXP];
#pragma unroll
    for (int e = 0; e < NEXP; ++e) acc[e] = 0.f;
    for (int d = lane; d < DMODEL; d += 32) {
        const float xv = xr[d];
        const float* wr = wsw + d * NEXP;
#pragma unroll
        for (int e = 0; e < NEXP; ++e) acc[e] += xv * wr[e];
    }
#pragma unroll
    for (int e = 0; e < NEXP; ++e)
        for (int off = 16; off; off >>= 1) acc[e] += __shfl_xor(acc[e], off, 32);
    if (lane == 0) {
        int best = 0;
        float bv = acc[0] + bsw[0];
#pragma unroll
        for (int e = 1; e < NEXP; ++e) {
            float v = acc[e] + bsw[e];
            if (v > bv) { bv = v; best = e; }   // strict > keeps first max (jnp.argmax)
        }
        routes[wid] = best;
    }
}

// --------------------------------------------------- deterministic ranking --
__global__ __launch_bounds__(256)
void slots_kernel(const int* __restrict__ routes, int* __restrict__ token_for_slot) {
    const int tid = threadIdx.x;
    for (int i = tid; i < NEXP * CAP; i += 256) token_for_slot[i] = -1;
    __syncthreads();
    const int e    = tid >> 5;   // wave id == expert id
    const int lane = tid & 31;
    int count = 0;
    for (int base = 0; base < T_TOK; base += 32) {
        const int  r     = routes[base + lane];
        const bool match = (r == e);
        const unsigned mlo = (unsigned)__ballot(match);
        const int prefix = __popc(mlo & ((1u << lane) - 1u));
        if (match) {
            const int pos = count + prefix;          // rank in token order
            if (pos < CAP) token_for_slot[e * CAP + pos] = base + lane;
        }
        count += __popc(mlo);
    }
}

// ---------------------------------------------------------- pass-through ----
__global__ __launch_bounds__(256)
void copy_kernel(const float4* __restrict__ in, float4* __restrict__ out, int n4) {
    for (int i = blockIdx.x * blockDim.x + threadIdx.x; i < n4;
         i += gridDim.x * blockDim.x)
        out[i] = in[i];
}

// ------------------------------------------------------- fused expert FFN ---
// grid.x = NEXP * (CAP/16) = 512 blocks; 256 threads (8 waves of wave32).
// Per block: 16 tokens. h[16x4096] lives entirely in LDS (bf16) — never in HBM.
// x-tile is gathered by the Tensor Data Mover (gather mode, 16-bit indices);
// empty slots use index 0xFFFF >= tensor_dim1 -> OOB reads return zero.
__global__ __launch_bounds__(256)
void expert_kernel(const float* __restrict__ x,
                   const float* __restrict__ w1, const float* __restrict__ b1,
                   const float* __restrict__ w2, const float* __restrict__ b2,
                   const int* __restrict__ token_for_slot,
                   float* __restrict__ out) {
    const int e    = blockIdx.x >> 6;
    const int tile = blockIdx.x & 63;
    const int tid  = threadIdx.x;
    const int wave = tid >> 5;
    const int lane = tid & 31;
    const int ln   = lane & 15;   // column (B/C/D) or row (A) index
    const int half = lane >> 4;   // lane-group per ISA 7.12.2 layout

    extern __shared__ __bf16 smem[];
    __bf16* xs  = smem;                                  // [16][XS_STRIDE] bf16
    __bf16* hs  = smem + 16 * XS_STRIDE;                 // [16][HS_STRIDE] bf16
    float*  xf  = (float*)(smem + 16 * XS_STRIDE + 16 * HS_STRIDE); // [16][1024] f32
    __shared__ int tok[16];

    const int slot0 = e * CAP + tile * 16;
    if (tid < 16) tok[tid] = token_for_slot[slot0 + tid];

    // ---- TDM gather: 16 token rows (f32) -> LDS staging, one DMA ----
    if (wave == 0) {
        // 16-bit row indices, 2 per dword (groups 2/3 of the D#)
        unsigned idx[16];
#pragma unroll
        for (int j = 0; j < 16; ++j) {
            const int t = token_for_slot[slot0 + j];           // uniform -> SMEM
            idx[j] = (t < 0) ? 0xFFFFu : (unsigned)t;          // OOB => zero row
        }
        const unsigned long long ga = (unsigned long long)(const void*)x;
        const unsigned lds_addr = (unsigned)(unsigned long long)(void*)xf;

        v4u g0;
        g0[0] = 0x80000001u;                       // count=1 | gather_mode
        g0[1] = lds_addr;                          // lds_addr [63:32]
        g0[2] = (unsigned)ga;                      // global_addr lo
        g0[3] = ((unsigned)(ga >> 32) & 0x01FFFFFFu) | 0x80000000u; // addr hi | type=2

        v8u g1;
        g1[0] = 0x00020000u;                       // data_size=4B, mask=0
        g1[1] = 0x04000000u;                       // tensor_dim0 = 1024 (bits 79:48)
        g1[2] = 0x40000000u;                       // tensor_dim1 = 16384 (bits 111:80)
        g1[3] = 0x04000000u;                       // tile_dim0 = 1024 (bits 127:112)
        g1[4] = 0x00000010u;                       // tile_dim1 = 16 valid indices
        g1[5] = 0x00000400u;                       // tensor_dim0_stride = 1024
        g1[6] = 0u;
        g1[7] = 0u;

        v4u g2, g3;
#pragma unroll
        for (int d = 0; d < 4; ++d) {
            g2[d] = idx[2 * d]     | (idx[2 * d + 1] << 16);
            g3[d] = idx[8 + 2 * d] | (idx[9 + 2 * d] << 16);
        }
        asm volatile("tensor_load_to_lds %0, %1, %2, %3"
                     :: "s"(g0), "s"(g1), "s"(g2), "s"(g3)
                     : "memory");
        __builtin_amdgcn_s_wait_tensorcnt(0);
    }
    __syncthreads();

    // ---- convert staged f32 rows -> bf16 xs tile ----
    {
        const int row = tid >> 4, seg = tid & 15;      // 16 threads per row
        const float* src = xf + row * DMODEL + seg * 64;
        __bf16*      dst = xs + row * XS_STRIDE + seg * 64;
#pragma unroll
        for (int i = 0; i < 64; ++i) dst[i] = (__bf16)src[i];
    }
    __syncthreads();

    const float* w1g = w1 + (size_t)e * DMODEL * HDIM;
    const float* b1g = b1 + e * HDIM;

    // ---- GEMM1: h[16,4096] = relu(x_tile @ W1 + b1), wave owns 512 cols ----
    for (int nt = 0; nt < 32; ++nt) {
        const int n0 = wave * 512 + nt * 16;
        v8f acc = {};
        for (int k0 = 0; k0 < DMODEL; k0 += 32) {
            if (k0 + 32 < DMODEL)   // prefetch next K-slice (global_prefetch_b8)
                __builtin_prefetch(&w1g[(size_t)(k0 + 32 + half * 8) * HDIM + n0 + ln], 0, 1);
            v16bf a, b;
#pragma unroll
            for (int j = 0; j < 16; ++j) {
                const int kk = (j & 7) + ((j >> 3) << 4) + half * 8; // ISA A-layout
                a[j] = xs[ln * XS_STRIDE + k0 + kk];
                b[j] = (__bf16)w1g[(size_t)(k0 + kk) * HDIM + n0 + ln];
            }
            acc = __builtin_amdgcn_wmma_f32_16x16x32_bf16(
                false, a, false, b, (short)0, acc, false, false);
        }
        const float bias = b1g[n0 + ln];
#pragma unroll
        for (int r = 0; r < 8; ++r) {                 // D layout: m = r + 8*half
            const int m = r + 8 * half;
            const float v = acc[r] + bias;
            hs[m * HS_STRIDE + n0 + ln] = (__bf16)(v > 0.f ? v : 0.f);
        }
    }
    __syncthreads();

    const float* w2g = w2 + (size_t)e * HDIM * DMODEL;
    const float* b2g = b2 + e * DMODEL;

    // ---- GEMM2: y[16,1024] = h @ W2 + b2, scatter rows to out ----
    for (int nt = 0; nt < 8; ++nt) {
        const int n0 = wave * 128 + nt * 16;
        v8f acc = {};
        for (int k0 = 0; k0 < HDIM; k0 += 32) {
            if (k0 + 32 < HDIM)
                __builtin_prefetch(&w2g[(size_t)(k0 + 32 + half * 8) * DMODEL + n0 + ln], 0, 1);
            v16bf a, b;
#pragma unroll
            for (int j = 0; j < 16; ++j) {
                const int kk = (j & 7) + ((j >> 3) << 4) + half * 8;
                a[j] = hs[ln * HS_STRIDE + k0 + kk];
                b[j] = (__bf16)w2g[(size_t)(k0 + kk) * DMODEL + n0 + ln];
            }
            acc = __builtin_amdgcn_wmma_f32_16x16x32_bf16(
                false, a, false, b, (short)0, acc, false, false);
        }
        const float bias = b2g[n0 + ln];
#pragma unroll
        for (int r = 0; r < 8; ++r) {
            const int m = r + 8 * half;
            const int token = tok[m];
            if (token >= 0)
                out[(size_t)token * DMODEL + n0 + ln] = acc[r] + bias;
        }
    }
}

// ------------------------------------------------------------------ launch --
extern "C" void kernel_launch(void* const* d_in, const int* in_sizes, int n_in,
                              void* d_out, int out_size, void* d_ws, size_t ws_size,
                              hipStream_t stream) {
    const float* x    = (const float*)d_in[0];
    const float* wsw  = (const float*)d_in[1];
    const float* bsw  = (const float*)d_in[2];
    const float* w1   = (const float*)d_in[3];
    const float* b1   = (const float*)d_in[4];
    const float* w2   = (const float*)d_in[5];
    const float* b2   = (const float*)d_in[6];
    float* out = (float*)d_out;

    int* routes         = (int*)d_ws;            // T_TOK ints
    int* token_for_slot = routes + T_TOK;        // NEXP*CAP ints

    // 1) router: 8 tokens (waves) per 256-thread block
    router_kernel<<<T_TOK / 8, 256, 0, stream>>>(x, wsw, bsw, routes);

    // 2) deterministic per-expert ranking + capacity truncation
    slots_kernel<<<1, 256, 0, stream>>>(routes, token_for_slot);

    // 3) pass-through base: out = x
    copy_kernel<<<2048, 256, 0, stream>>>((const float4*)x, (float4*)out,
                                          (T_TOK * DMODEL) / 4);

    // 4) fused expert FFN (bf16 WMMA + TDM gather; h kept entirely in LDS)
    const size_t lds_bytes =
        (size_t)(16 * XS_STRIDE + 16 * HS_STRIDE) * sizeof(__bf16)  // bf16 tiles
        + (size_t)16 * DMODEL * sizeof(float);                      // f32 staging
    expert_kernel<<<NEXP * (CAP / 16), 256, lds_bytes, stream>>>(
        x, w1, b1, w2, b2, token_for_slot, out);
}
